// CBOW_56530359550882
// MI455X (gfx1250) — matrix operand order repaired and checked by reading, not compiled, not run on hardware
//
#include <hip/hip_runtime.h>

#define DIM      128
#define CTX      10
#define NEG      10
#define ROWS     11            // target + 10 negatives
#define WPB      8             // waves per block (256 threads, 8 wave32)
#define BLOCK    256
#define ROW_U32  68            // 128 f16 = 64 dwords, +4 pad -> conflict-free WMMA A reads
#define CM_U32   32            // 128 f16 context-mean vector
#define WAVE_U32 (ROWS * ROW_U32 + CM_U32)   // 780 dwords per wave

typedef __attribute__((ext_vector_type(16))) _Float16 v16h;
typedef __attribute__((ext_vector_type(8)))  float    v8f;
typedef __attribute__((ext_vector_type(2)))  __fp16   v2fp16;   // return type of cvt_pkrtz

union HCast { struct { uint4 lo, hi; } u; v16h v; };

__device__ __forceinline__ uint32_t pack_f16x2(float a, float b) {
#if __has_builtin(__builtin_amdgcn_cvt_pkrtz)
  union { v2fp16 h; uint32_t u; } c;
  c.h = __builtin_amdgcn_cvt_pkrtz(a, b);     // single v_cvt_pk_f16_f32
  return c.u;
#else
  union { _Float16 h[2]; uint32_t u; } c;
  c.h[0] = (_Float16)a; c.h[1] = (_Float16)b;
  return c.u;
#endif
}

// numerically stable log(sigmoid(x)) = min(x,0) - log(1 + exp(-|x|)).
// log argument is confined to (1,2] so the hardware v_log_f32 (__logf) is
// accurate there; avoids the huge precise log1pf expansion seen in round 1.
__device__ __forceinline__ float log_sigmoid(float x) {
  return fminf(x, 0.0f) - __logf(1.0f + __expf(-fabsf(x)));
}

__global__ __launch_bounds__(BLOCK) void cbow_loss_kernel(
    const float* __restrict__ W_embed, const float* __restrict__ W_out,
    const int* __restrict__ context_ids, const int* __restrict__ target_ids,
    const int* __restrict__ neg_ids, float* __restrict__ block_partial,
    int Btotal)
{
  __shared__ __align__(16) uint32_t lds[WPB * WAVE_U32];
  __shared__ float s_part[WPB];

  const int lane = threadIdx.x & 31;
  const int wave = threadIdx.x >> 5;
  const int braw = blockIdx.x * WPB + wave;
  const int b    = braw < Btotal ? braw : (Btotal - 1);   // clamp; masked below
  const float valid = (braw < Btotal) ? 1.0f : 0.0f;

  // ---- wave-uniform sample ids ----
  int rid[ROWS];
  rid[0] = target_ids[b];
  #pragma unroll
  for (int n = 0; n < NEG; ++n) rid[1 + n] = neg_ids[b * NEG + n];

  // ---- prefetch the 11 W_out rows (global_prefetch_b8) while context gathers run
  #pragma unroll
  for (int r = 0; r < ROWS; ++r)
    __builtin_prefetch(W_out + (size_t)rid[r] * DIM + lane * 4, 0, 1);

  // ---- context mean: lane owns dims [4*lane, 4*lane+3], 512B coalesced per row
  float4 cm = make_float4(0.f, 0.f, 0.f, 0.f);
  #pragma unroll
  for (int c = 0; c < CTX; ++c) {
    const int id = context_ids[b * CTX + c];
    const float4 e = *(const float4*)(W_embed + (size_t)id * DIM + lane * 4);
    cm.x += e.x; cm.y += e.y; cm.z += e.z; cm.w += e.w;
  }
  const float invc = 1.0f / (float)CTX;
  cm.x *= invc; cm.y *= invc; cm.z *= invc; cm.w *= invc;

  // ---- stage 11 output rows + cm into LDS as f16 (pitch-68 padded)
  uint32_t* wb = lds + wave * WAVE_U32;
  #pragma unroll
  for (int r = 0; r < ROWS; ++r) {
    const float4 e = *(const float4*)(W_out + (size_t)rid[r] * DIM + lane * 4);
    uint2 p; p.x = pack_f16x2(e.x, e.y); p.y = pack_f16x2(e.z, e.w);
    *(uint2*)(wb + r * ROW_U32 + lane * 2) = p;
  }
  {
    uint2 p; p.x = pack_f16x2(cm.x, cm.y); p.y = pack_f16x2(cm.z, cm.w);
    *(uint2*)(wb + ROWS * ROW_U32 + lane * 2) = p;
  }
  __syncthreads();

  // ---- scores via V_WMMA_F32_16X16X32_F16, K accumulated over 4 steps (=128)
  // A: 16x32 f16 tile -> lane L supplies row (L&15); ISA layout:
  //   lanes 0-15 : VGPR0-3 = K 0..7,  VGPR4-7 = K 16..23
  //   lanes16-31 : VGPR0-3 = K 8..15, VGPR4-7 = K 24..31
  // B: 32x16 f16, cm replicated across all N columns (same K order as A:
  //   lanes 0-15: K0..7,16..23; lanes 16-31: K8..15,24..31) -> same-address LDS broadcasts.
  const int hi = lane >> 4;
  int arow = lane & 15; if (arow >= ROWS) arow = 0;   // alias pad rows; D rows 11..15 unused
  const uint32_t* ap = wb + arow * ROW_U32;
  const uint32_t* bp = wb + ROWS * ROW_U32;

  v8f acc = {0.f, 0.f, 0.f, 0.f, 0.f, 0.f, 0.f, 0.f};
  #pragma unroll
  for (int s = 0; s < 4; ++s) {
    HCast A, Bm;
    A.u.lo  = *(const uint4*)(ap + 16 * s + 4 * hi);      // K 0..7   (+8 for hi lanes)
    A.u.hi  = *(const uint4*)(ap + 16 * s + 4 * hi + 8);  // K 16..23 (+8 for hi lanes)
    Bm.u.lo = *(const uint4*)(bp + 16 * s + 4 * hi);      // cm halves, broadcast
    Bm.u.hi = *(const uint4*)(bp + 16 * s + 4 * hi + 8);
    acc = __builtin_amdgcn_wmma_f32_16x16x32_f16(
        /*neg_a=*/false, A.v, /*neg_b=*/false, Bm.v,
        /*c_mod=*/(short)0, acc, /*reuse_a=*/false, /*reuse_b=*/false);
  }

  // ---- extract scores: D layout => row i<8 in acc[i] (lanes 0-15), row 8+j in acc[j] (lanes 16-31)
  float x0  = __shfl(acc[0], 0, 32);   // pos score (target)
  float x1  = __shfl(acc[1], 0, 32);
  float x2  = __shfl(acc[2], 0, 32);
  float x3  = __shfl(acc[3], 0, 32);
  float x4  = __shfl(acc[4], 0, 32);
  float x5  = __shfl(acc[5], 0, 32);
  float x6  = __shfl(acc[6], 0, 32);
  float x7  = __shfl(acc[7], 0, 32);
  float x8  = __shfl(acc[0], 16, 32);
  float x9  = __shfl(acc[1], 16, 32);
  float x10 = __shfl(acc[2], 16, 32);

  float lb = log_sigmoid(x0)
           + log_sigmoid(-x1) + log_sigmoid(-x2) + log_sigmoid(-x3)
           + log_sigmoid(-x4) + log_sigmoid(-x5) + log_sigmoid(-x6)
           + log_sigmoid(-x7) + log_sigmoid(-x8) + log_sigmoid(-x9)
           + log_sigmoid(-x10);
  lb *= valid;

  if (lane == 0) s_part[wave] = lb;
  __syncthreads();
  if (threadIdx.x == 0) {
    float s = 0.f;
    #pragma unroll
    for (int w = 0; w < WPB; ++w) s += s_part[w];
    block_partial[blockIdx.x] = s;
  }
}

__global__ __launch_bounds__(BLOCK) void cbow_reduce_kernel(
    const float* __restrict__ part, int n, float* __restrict__ out, float invB)
{
  __shared__ float s[BLOCK];
  float a = 0.f;
  for (int i = threadIdx.x; i < n; i += BLOCK) a += part[i];   // fixed order -> deterministic
  s[threadIdx.x] = a;
  __syncthreads();
  for (int st = BLOCK / 2; st > 0; st >>= 1) {
    if ((int)threadIdx.x < st) s[threadIdx.x] += s[threadIdx.x + st];
    __syncthreads();
  }
  if (threadIdx.x == 0) out[0] = -s[0] * invB;
}

extern "C" void kernel_launch(void* const* d_in, const int* in_sizes, int n_in,
                              void* d_out, int out_size, void* d_ws, size_t ws_size,
                              hipStream_t stream)
{
  const float* W_embed     = (const float*)d_in[0];
  const float* W_out       = (const float*)d_in[1];
  const int*   context_ids = (const int*)d_in[2];
  const int*   target_ids  = (const int*)d_in[3];
  const int*   neg_ids     = (const int*)d_in[4];

  const int B    = in_sizes[3];                // 16384
  const int grid = (B + WPB - 1) / WPB;        // 2048 blocks, 1 wave per sample
  float* part = (float*)d_ws;                  // grid * 4 bytes of scratch

  cbow_loss_kernel<<<grid, BLOCK, 0, stream>>>(
      W_embed, W_out, context_ids, target_ids, neg_ids, part, B);
  cbow_reduce_kernel<<<1, BLOCK, 0, stream>>>(
      part, grid, (float*)d_out, 1.0f / (float)B);
}
